// CausalSelfAttention_31258771980754
// MI455X (gfx1250) — matrix operand order repaired and checked by reading, not compiled, not run on hardware
//
#include <hip/hip_runtime.h>
#include <hip/hip_bf16.h>

typedef __attribute__((ext_vector_type(16))) _Float16 v16h;
typedef __attribute__((ext_vector_type(8)))  _Float16 v8h;
typedef __attribute__((ext_vector_type(8)))  float    v8f;
typedef __attribute__((ext_vector_type(4)))  unsigned int v4u;
typedef __attribute__((ext_vector_type(8)))  int      v8i;
typedef __attribute__((ext_vector_type(4)))  int      v4i;

#define WMMA_F32_F16(A_, B_, C_) \
  __builtin_amdgcn_wmma_f32_16x16x32_f16(false, (A_), false, (B_), (short)0, (C_), false, false)

// ---------------------------------------------------------------------------
// Operand loaders (CDNA5 WMMA wave32 VGPR layouts, CDNA5 ISA 7.12.2)
// ---------------------------------------------------------------------------

// A operand: 16x32 (MxK) f16 tile from a K-contiguous (row-major) matrix.
// lanes 0-15 : row M = lane,    elems 0..7 = K 0..7,   elems 8..15 = K 16..23
// lanes 16-31: row M = lane-16, elems 0..7 = K 8..15,  elems 8..15 = K 24..31
__device__ __forceinline__ v16h load_a_tile(const _Float16* base, long ld, int lane) {
  const int r  = lane & 15;
  const int ko = (lane & 16) ? 8 : 0;
  const _Float16* p = base + (long)r * ld + ko;
  v8h lo = *(const v8h*)(p);
  v8h hi = *(const v8h*)(p + 16);
  v16h out;
#pragma unroll
  for (int i = 0; i < 8; ++i) { out[i] = lo[i]; out[i + 8] = hi[i]; }
  return out;
}

// B operand: 32x16 (KxN) f16 tile. Source matrix is N rows of K-contiguous
// data (i.e. we compute X * W^T with both K-major).
// lanes 0-15 : col N = lane,    elems 0..7 = K 0..7,   elems 8..15 = K 16..23
// lanes 16-31: col N = lane-16, elems 0..7 = K 8..15,  elems 8..15 = K 24..31
__device__ __forceinline__ v16h load_b_tile(const _Float16* base, long ld, int lane) {
  const int c  = lane & 15;
  const int ko = (lane & 16) ? 8 : 0;
  const _Float16* p = base + (long)c * ld + ko;
  v8h lo = *(const v8h*)(p);
  v8h hi = *(const v8h*)(p + 16);
  v16h out;
#pragma unroll
  for (int i = 0; i < 8; ++i) { out[i] = lo[i]; out[i + 8] = hi[i]; }
  return out;
}

// ---------------------------------------------------------------------------
// Tensor Data Mover: 2D tile load Global -> LDS (CDNA5 ISA)
//   tile_d0 = elements per row, tile_d1 = rows, stride0 = row stride (elements)
//   f16 elements (data_size code 1 = 2 bytes). Tracked with TENSORcnt.
// ---------------------------------------------------------------------------
__device__ __forceinline__ unsigned lds_offset_of(const void* p) {
  // generic LDS address: high bits = shared aperture, addr[31:0] = LDS offset
  return (unsigned)(unsigned long long)p;
}

__device__ __forceinline__ void tdm_load_2d_f16(unsigned lds_addr, const void* gaddr,
                                                unsigned tile_d0, unsigned tile_d1,
                                                unsigned long stride0) {
  const unsigned long ga = (unsigned long)gaddr;
  v4u g0;
  g0[0] = 1u;                                                  // count=1, user D#
  g0[1] = lds_addr;                                            // lds_addr[31:0]
  g0[2] = (unsigned)(ga & 0xffffffffu);                        // global_addr[31:0]
  g0[3] = (unsigned)((ga >> 32) & 0x1ffffffu) | (2u << 30);    // addr[56:32] | type=2
  v8i g1;
  g1[0] = (int)(1u << 16);                                     // wg_mask=0, data_size=1 (2B)
  g1[1] = (int)((tile_d0 & 0xffffu) << 16);                    // tensor_dim0[15:0]
  g1[2] = (int)(((tile_d0 >> 16) & 0xffffu) |                  // tensor_dim0[31:16]
                ((tile_d1 & 0xffffu) << 16));                  // tensor_dim1[15:0]
  g1[3] = (int)(((tile_d1 >> 16) & 0xffffu) |                  // tensor_dim1[31:16]
                ((tile_d0 & 0xffffu) << 16));                  // tile_dim0
  g1[4] = (int)(tile_d1 & 0xffffu);                            // tile_dim1, tile_dim2=0
  g1[5] = (int)(unsigned)(stride0 & 0xffffffffu);              // dim0_stride[31:0]
  g1[6] = (int)(unsigned)((stride0 >> 32) & 0xffffu);          // dim0_stride[47:32]
  g1[7] = 0;                                                   // dim1_stride unused (2D tile)
  v4i z4 = {0, 0, 0, 0};
#if __clang_major__ >= 23
  v8i z8 = {0, 0, 0, 0, 0, 0, 0, 0};
  __builtin_amdgcn_tensor_load_to_lds(g0, g1, z4, z4, z8, 0);
#else
  __builtin_amdgcn_tensor_load_to_lds(g0, g1, z4, z4, 0);
#endif
}

// ---------------------------------------------------------------------------
// Kernel 1: f32 -> f16 convert
// ---------------------------------------------------------------------------
__global__ void f32_to_f16_kernel(const float* __restrict__ in,
                                  _Float16* __restrict__ out, long n) {
  long i = (long)blockIdx.x * blockDim.x + threadIdx.x;
  if (i < n) out[i] = (_Float16)in[i];
}

// ---------------------------------------------------------------------------
// Kernel 2/5: WMMA GEMM  C[M,N] = A[M,K] * Bw[N,K]^T   (both f16, K-major)
// One wave per block; each wave owns a 64x32 output tile (4x2 C tiles).
// 8 v_wmma_f32_16x16x32_f16 per K-step of 32.
// ---------------------------------------------------------------------------
template <typename OutT>
__global__ void __launch_bounds__(32)
gemm_wmma_kernel(const _Float16* __restrict__ A, const _Float16* __restrict__ Bw,
                 OutT* __restrict__ C, int M, int N, int K) {
  const int lane = threadIdx.x;
  const int m0 = blockIdx.x * 64;
  const int n0 = blockIdx.y * 32;

  v8f acc[4][2] = {};

  for (int k = 0; k < K; k += 32) {
    v16h a[4], b[2];
#pragma unroll
    for (int j = 0; j < 2; ++j)
      b[j] = load_b_tile(Bw + (long)(n0 + 16 * j) * K + k, K, lane);
#pragma unroll
    for (int i = 0; i < 4; ++i)
      a[i] = load_a_tile(A + (long)(m0 + 16 * i) * K + k, K, lane);

    if (k + 32 < K) {  // gfx1250 global_prefetch_b8 hint for the next K chunk
      __builtin_prefetch(A + (long)(m0 + lane) * K + k + 32, 0, 0);
      __builtin_prefetch(Bw + (long)(n0 + (lane & 31)) * K + k + 32, 0, 0);
    }

#pragma unroll
    for (int i = 0; i < 4; ++i)
#pragma unroll
      for (int j = 0; j < 2; ++j)
        acc[i][j] = WMMA_F32_F16(a[i], b[j], acc[i][j]);
  }

  // C/D layout: lane = col within 16 (upper lane half -> rows +8), vgpr = row
  const int n = lane & 15;
  const int rbase = (lane & 16) ? 8 : 0;
#pragma unroll
  for (int i = 0; i < 4; ++i)
#pragma unroll
    for (int j = 0; j < 2; ++j)
#pragma unroll
      for (int r = 0; r < 8; ++r) {
        long m = m0 + 16 * i + rbase + r;
        C[m * (long)N + n0 + 16 * j + n] = (OutT)acc[i][j][r];
      }
}

// ---------------------------------------------------------------------------
// Kernel 3: RoPE + head split. qkv is (B,S,3*1024) f16.
//   Q -> (B,H,S,64) f16, pre-scaled by 1/8 (= hd^-0.5)
//   K -> (B,H,S,64) f16
//   V -> (B,H,64,S) f16  (transposed so P@V B-operand reads are contiguous)
// One thread per (b,h,s).
// ---------------------------------------------------------------------------
__global__ void rope_reshape_kernel(const _Float16* __restrict__ qkv,
                                    _Float16* __restrict__ Qo,
                                    _Float16* __restrict__ Ko,
                                    _Float16* __restrict__ Vt,
                                    int B, int H, int S) {
  long idx = (long)blockIdx.x * blockDim.x + threadIdx.x;  // over B*H*S
  if (idx >= (long)B * H * S) return;
  int s = (int)(idx % S);
  int h = (int)((idx / S) % H);
  int b = (int)(idx / ((long)S * H));

  const _Float16* base = qkv + ((long)b * S + s) * 3072;
  const _Float16* q = base + h * 64;
  const _Float16* k = base + 1024 + h * 64;
  const _Float16* v = base + 2048 + h * 64;

  long row = (((long)b * H + h) * S + s) * 64;

  const float LN1E4_OVER_HALF = 9.210340371976184f / 32.0f;  // ln(10000)/32
#pragma unroll 4
  for (int j = 0; j < 32; ++j) {
    float freq = __expf(-(float)j * LN1E4_OVER_HALF);
    float ang = (float)s * freq;
    float sn, cs;
    __sincosf(ang, &sn, &cs);
    float q1 = (float)q[j], q2 = (float)q[j + 32];
    float k1 = (float)k[j], k2 = (float)k[j + 32];
    Qo[row + j]      = (_Float16)(0.125f * (q1 * cs - q2 * sn));
    Qo[row + j + 32] = (_Float16)(0.125f * (q2 * cs + q1 * sn));
    Ko[row + j]      = (_Float16)(k1 * cs - k2 * sn);
    Ko[row + j + 32] = (_Float16)(k2 * cs + k1 * sn);
  }

  long vt = ((long)b * H + h) * 64 * S + s;
#pragma unroll 8
  for (int d = 0; d < 64; ++d) Vt[vt + (long)d * S] = v[d];
}

// ---------------------------------------------------------------------------
// Kernel 4: causal flash attention, one wave per 16-query tile per (b,h).
//   K/V chunks are DMA'd Global->LDS by the Tensor Data Mover, double-buffered
//   against compute with TENSORcnt (issue chunk c+1, s_wait_tensorcnt 2,
//   compute chunk c from LDS).
//   scores = Q(16x64) @ K^T       (2x2 WMMAs per 32-key chunk, operands ds_load)
//   online softmax in C layout (rows spread over 16-lane halves -> shfl_xor)
//   P relayout C->A via LDS bounce (LDS is in-order within a wave)
//   O += P(16x32) @ V(32x64)      (4 WMMAs into 4 C tiles)
// ---------------------------------------------------------------------------
__global__ void __launch_bounds__(32)
flash_attn_kernel(const _Float16* __restrict__ Q, const _Float16* __restrict__ K,
                  const _Float16* __restrict__ Vt, _Float16* __restrict__ ctx,
                  int B, int H, int S) {
  __shared__ __align__(16) _Float16 kbuf[2][32 * 64];  // [chunk][key][d]
  __shared__ __align__(16) _Float16 vbuf[2][64 * 32];  // [chunk][d][key]
  __shared__ __align__(16) _Float16 psh[16 * 32];      // P relayout bounce

  const int lane = threadIdx.x;
  const int q0 = blockIdx.x * 16;
  const int bh = blockIdx.y;
  const int b = bh / H, h = bh % H;

  const _Float16* Qb = Q + ((long)bh * S + q0) * 64;
  const _Float16* Kb = K + (long)bh * S * 64;
  const _Float16* Vb = Vt + (long)bh * 64 * S;

  const v16h aq0 = load_a_tile(Qb, 64, lane);        // d = 0..31
  const v16h aq1 = load_a_tile(Qb + 32, 64, lane);   // d = 32..63

  v8f o[4] = {};
  float mrow[8], lrow[8];
#pragma unroll
  for (int r = 0; r < 8; ++r) { mrow[r] = -1e30f; lrow[r] = 0.0f; }

  const int n = lane & 15;
  const int rbase = (lane & 16) ? 8 : 0;

  const int nchunks = (q0 + 16 + 31) / 32;

  // TDM issue for chunk c into buffer (c & 1): K tile 32 rows x 64 f16
  // (contiguous), V tile 64 rows x 32 f16 with row stride S.
  auto issue_chunk = [&](int c) {
    const int k0 = c * 32;
    const int sl = c & 1;
    tdm_load_2d_f16(lds_offset_of(&kbuf[sl][0]), Kb + (long)k0 * 64, 64, 32, 64);
    tdm_load_2d_f16(lds_offset_of(&vbuf[sl][0]), Vb + k0, 32, 64, (unsigned long)S);
  };

  issue_chunk(0);

  for (int c = 0; c < nchunks; ++c) {
    const int k0 = c * 32;
    const int sl = c & 1;
    const bool has_next = (c + 1 < nchunks);
    if (has_next) issue_chunk(c + 1);

    // consume chunk c: its 2 TDM ops done once TENSORcnt <= (next ? 2 : 0)
    if (has_next) __builtin_amdgcn_s_wait_tensorcnt(2);
    else          __builtin_amdgcn_s_wait_tensorcnt(0);
    asm volatile("" ::: "memory");

    const _Float16* kb = &kbuf[sl][0];
    const _Float16* vb = &vbuf[sl][0];

    // ---- scores: 16 queries x 32 keys (two 16-key groups), operands from LDS
    v16h bk0a = load_b_tile(kb, 64, lane);             // g0, d 0..31
    v16h bk0b = load_b_tile(kb + 32, 64, lane);        // g0, d 32..63
    v16h bk1a = load_b_tile(kb + 16 * 64, 64, lane);   // g1, d 0..31
    v16h bk1b = load_b_tile(kb + 16 * 64 + 32, 64, lane);

    v8f sc0 = {}, sc1 = {};
    sc0 = WMMA_F32_F16(aq0, bk0a, sc0);
    sc0 = WMMA_F32_F16(aq1, bk0b, sc0);
    sc1 = WMMA_F32_F16(aq0, bk1a, sc1);
    sc1 = WMMA_F32_F16(aq1, bk1b, sc1);

    // ---- causal mask + online softmax ----
#pragma unroll
    for (int r = 0; r < 8; ++r) {
      const int m = q0 + rbase + r;
      float s0 = sc0[r]; if (k0 + n > m)      s0 = -1e30f;
      float s1 = sc1[r]; if (k0 + 16 + n > m) s1 = -1e30f;

      float vmax = fmaxf(s0, s1);
#pragma unroll
      for (int msk = 8; msk >= 1; msk >>= 1)
        vmax = fmaxf(vmax, __shfl_xor(vmax, msk, 32));

      const float newm = fmaxf(mrow[r], vmax);
      const float corr = __expf(mrow[r] - newm);
      mrow[r] = newm;

      const float p0 = __expf(s0 - newm);
      const float p1 = __expf(s1 - newm);
      psh[(rbase + r) * 32 + n]      = (_Float16)p0;
      psh[(rbase + r) * 32 + 16 + n] = (_Float16)p1;

      float rs = p0 + p1;
#pragma unroll
      for (int msk = 8; msk >= 1; msk >>= 1)
        rs += __shfl_xor(rs, msk, 32);
      lrow[r] = lrow[r] * corr + rs;

#pragma unroll
      for (int t = 0; t < 4; ++t) o[t][r] *= corr;
    }

    // ---- O += P @ V : P as A operand (LDS bounce), V chunk from LDS ----
    const v16h pa = load_a_tile(psh, 32, lane);
#pragma unroll
    for (int t = 0; t < 4; ++t) {
      v16h bv = load_b_tile(vb + 16 * t * 32, 32, lane);
      o[t] = WMMA_F32_F16(pa, bv, o[t]);
    }
  }

  // ---- normalize and store context in (B,S,1024) f16 ----
  const long obase = (long)b * S * 1024 + (long)h * 64;
#pragma unroll
  for (int r = 0; r < 8; ++r) {
    const int m = q0 + rbase + r;
    const float inv = 1.0f / lrow[r];
    const long row = obase + (long)m * 1024;
#pragma unroll
    for (int t = 0; t < 4; ++t)
      ctx[row + 16 * t + n] = (_Float16)(o[t][r] * inv);
  }
}

// ---------------------------------------------------------------------------
// Host launch
// ---------------------------------------------------------------------------
extern "C" void kernel_launch(void* const* d_in, const int* in_sizes, int n_in,
                              void* d_out, int out_size, void* d_ws, size_t ws_size,
                              hipStream_t stream) {
  const int B = 2, S = 2048, D = 1024, H = 16;
  const long M = (long)B * S;  // 4096

  const float* x     = (const float*)d_in[0];
  const float* w_qkv = (const float*)d_in[1];
  const float* w_out = (const float*)d_in[2];
  float* out = (float*)d_out;

  char* ws = (char*)d_ws;
  // workspace layout (bytes); ctx overlays xh (dead after QKV GEMM)
  _Float16* xh    = (_Float16*)(ws + 0);           //  8 MiB (B,S,D)
  _Float16* ctx   = (_Float16*)(ws + 0);           //  8 MiB (B,S,D) overlay
  _Float16* wqkvh = (_Float16*)(ws + 8388608);     //  6 MiB (3D,D)
  _Float16* wouth = (_Float16*)(ws + 14680064);    //  2 MiB (D,D)
  _Float16* qkvh  = (_Float16*)(ws + 16777216);    // 24 MiB (B,S,3D)
  _Float16* Qh    = (_Float16*)(ws + 41943040);    //  8 MiB (B,H,S,64)
  _Float16* Kh    = (_Float16*)(ws + 50331648);    //  8 MiB (B,H,S,64)
  _Float16* Vt    = (_Float16*)(ws + 58720256);    //  8 MiB (B,H,64,S)
  if (ws_size < 67108864) return;

  // 1) downconvert inputs to f16
  long nx = M * D;
  f32_to_f16_kernel<<<(nx + 255) / 256, 256, 0, stream>>>(x, xh, nx);
  f32_to_f16_kernel<<<(3L * D * D + 255) / 256, 256, 0, stream>>>(w_qkv, wqkvh, 3L * D * D);
  f32_to_f16_kernel<<<((long)D * D + 255) / 256, 256, 0, stream>>>(w_out, wouth, (long)D * D);

  // 2) QKV projection: (4096 x 1024) * (3072 x 1024)^T -> f16 (4096 x 3072)
  gemm_wmma_kernel<_Float16><<<dim3(M / 64, (3 * D) / 32), 32, 0, stream>>>(
      xh, wqkvh, qkvh, (int)M, 3 * D, D);

  // 3) RoPE + head split (+ V transpose)
  rope_reshape_kernel<<<(int)(((long)B * H * S + 255) / 256), 256, 0, stream>>>(
      qkvh, Qh, Kh, Vt, B, H, S);

  // 4) causal flash attention (TDM-staged K/V, double-buffered on TENSORcnt)
  flash_attn_kernel<<<dim3(S / 16, B * H), 32, 0, stream>>>(Qh, Kh, Vt, ctx, B, H, S);

  // 5) output projection: (4096 x 1024) * (1024 x 1024)^T -> f32 out
  gemm_wmma_kernel<float><<<dim3(M / 64, D / 32), 32, 0, stream>>>(
      ctx, wouth, out, (int)M, D, D);
}
